// SPVCNNDET_17068200035067
// MI455X (gfx1250) — matrix operand order repaired and checked by reading, not compiled, not run on hardware
//
#include <hip/hip_runtime.h>

typedef __bf16 bf16;
typedef __attribute__((ext_vector_type(16))) __bf16 v16bf;
typedef __attribute__((ext_vector_type(8)))  __bf16 v8bf;
typedef __attribute__((ext_vector_type(8)))  float  v8f;

#define BATCH 4
#define U3D 200
#define V3D 176
#define Z3D 5
#define CBEV 256

#if defined(__AMDGCN__) && __has_builtin(__builtin_amdgcn_global_load_async_to_lds_b128)
#define HAVE_ASYNC_LDS 1
#else
#define HAVE_ASYNC_LDS 0
#endif

#if HAVE_ASYNC_LDS
typedef int v4i_t __attribute__((vector_size(16)));
typedef __attribute__((address_space(1))) v4i_t* gvec4p;   // global int4*
typedef __attribute__((address_space(3))) v4i_t* lvec4p;   // LDS int4*
#if __has_builtin(__builtin_amdgcn_s_wait_asynccnt)
#define WAIT_ASYNC(n) __builtin_amdgcn_s_wait_asynccnt(n)
#else
#define WAIT_ASYNC(n) asm volatile("s_wait_asynccnt %0" ::"n"(n) : "memory")
#endif
#endif

// ---------------------------------------------------------------------------
// Sparse conv: Y[n, co] = sum_k sum_ci Xpad[rb[k][n], ci] * W[k][ci][co]
// 256-thread block = 8 waves; each wave owns a 16-row x COUT tile (128 rows
// per block).  Per kernel offset k the W[k] slice is staged into LDS once per
// block (async double-buffered via GLOBAL_LOAD_ASYNC_TO_LDS_B128 + ASYNCcnt),
// then consumed by all 8 waves via per-element 16-bit LDS reads.
//   A layout (16-bit A 16x32, ISA 7.12.2): lane<16 -> row m=lane, K in
//     {kc..kc+7, kc+16..kc+23}; lane>=16 -> row m=lane-16, K in
//     {kc+8..kc+15, kc+24..kc+31}.
//   B layout (32x16 16-bit): lane L holds column N=L%16, element j -> K in the same order as A.
//   D layout: VGPR r, lanes 0-15 -> (M=r, N=lane); lanes 16-31 -> (M=r+8).
// ---------------------------------------------------------------------------
template<int CINP, int COUT>
__global__ __launch_bounds__(256)
void spconv_wmma(const bf16* __restrict__ X, const int* __restrict__ rb, int numK,
                 const bf16* __restrict__ W, float* __restrict__ Y,
                 int Nin, int Nout)
{
    constexpr int NT     = COUT / 16;
    constexpr int PITCH  = COUT + 8;                 // keeps 16B row alignment
    constexpr int CHUNKS = (CINP * COUT) / 8;        // 16-byte chunks per slice
    constexpr int CPT    = (CHUNKS + 255) / 256;     // async loads per thread

#if HAVE_ASYNC_LDS
    __shared__ bf16 smW[2][CINP * PITCH];
#else
    __shared__ bf16 smW[1][CINP * PITCH];
#endif

    const int tid   = threadIdx.x;
    const int lane  = tid & 31;
    const int wave  = tid >> 5;
    const int row0  = blockIdx.x * 128 + wave * 16;
    const int m     = lane & 15;
    const int khalf = lane >> 4;

    v8f acc[NT];
#pragma unroll
    for (int t = 0; t < NT; ++t) {
        v8f z;
#pragma unroll
        for (int j = 0; j < 8; ++j) z[j] = 0.0f;
        acc[t] = z;
    }

    const int r = row0 + m;

#if HAVE_ASYNC_LDS
    // ---- async double-buffered staging: stage k=0 ----
    {
        const bf16* wk0 = W;
        for (int c = tid; c < CHUNKS; c += 256) {
            int row = (c * 8) / COUT, col = (c * 8) % COUT;
            __builtin_amdgcn_global_load_async_to_lds_b128(
                (gvec4p)(wk0 + (long)row * COUT + col),
                (lvec4p)(&smW[0][row * PITCH + col]),
                0, 0);
        }
    }
#endif

    for (int k = 0; k < numK; ++k) {
#if HAVE_ASYNC_LDS
        if (k + 1 < numK) {
            const bf16* wkn = W + (long)(k + 1) * (CINP * COUT);
            const int buf = (k + 1) & 1;
            for (int c = tid; c < CHUNKS; c += 256) {
                int row = (c * 8) / COUT, col = (c * 8) % COUT;
                __builtin_amdgcn_global_load_async_to_lds_b128(
                    (gvec4p)(wkn + (long)row * COUT + col),
                    (lvec4p)(&smW[buf][row * PITCH + col]),
                    0, 0);
            }
            WAIT_ASYNC(CPT);   // stage k complete; stage k+1 still in flight
        } else {
            WAIT_ASYNC(0);
        }
        __syncthreads();
        const bf16* bb = &smW[k & 1][0];
#else
        // ---- synchronous staging fallback ----
        __syncthreads();
        {
            const bf16* wk = W + (long)k * (CINP * COUT);
            for (int c = tid; c < CHUNKS; c += 256) {
                int row = (c * 8) / COUT, col = (c * 8) % COUT;
                v8bf v = *(const v8bf*)(wk + (long)row * COUT + col);
                *(v8bf*)(&smW[0][row * PITCH + col]) = v;
            }
        }
        __syncthreads();
        const bf16* bb = &smW[0][0];
#endif

        int idx;
        if (rb) idx = (r < Nout) ? rb[(long)k * Nout + r] : Nin;
        else    idx = (r < Nout) ? r : Nin;
        const bool valid = ((unsigned)idx < (unsigned)Nin);
        const bf16* xrow = X + (long)idx * CINP;

#pragma unroll
        for (int kc = 0; kc < CINP; kc += 32) {
            v16bf a;
            if (valid) {
                v8bf lo = *(const v8bf*)(xrow + kc + khalf * 8);
                v8bf hi = *(const v8bf*)(xrow + kc + 16 + khalf * 8);
                a = __builtin_shufflevector(lo, hi, 0,1,2,3,4,5,6,7,
                                                    8,9,10,11,12,13,14,15);
            } else {
#pragma unroll
                for (int j = 0; j < 16; ++j) a[j] = (bf16)0.0f;
            }
            const bf16* brow = bb + kc * PITCH + m;
#pragma unroll
            for (int t = 0; t < NT; ++t) {
                v16bf bmat;
#pragma unroll
                for (int j = 0; j < 16; ++j)
                    bmat[j] = brow[((j < 8) ? (khalf * 8 + j) : (16 + khalf * 8 + (j - 8))) * PITCH + t * 16];
                acc[t] = __builtin_amdgcn_wmma_f32_16x16x32_bf16(
                    false, a, false, bmat, (short)0, acc[t], false, false);
            }
        }
#if HAVE_ASYNC_LDS
        __syncthreads();   // all waves done reading smW[k&1] before overwrite
#endif
    }

    const int n  = lane & 15;
    const int mo = khalf * 8;
#pragma unroll
    for (int t = 0; t < NT; ++t) {
#pragma unroll
        for (int rr = 0; rr < 8; ++rr) {
            int row = row0 + mo + rr;
            if (row < Nout) Y[(long)row * COUT + t * 16 + n] = acc[t][rr];
        }
    }
}

// f32 weights [numK][Cin][Cout] -> bf16 [numK][CinPad][Cout], zero pad rows
__global__ void conv_w_cvt(const float* __restrict__ W, bf16* __restrict__ Wb,
                           int numK, int Cin, int CinPad, int Cout)
{
    long i = (long)blockIdx.x * blockDim.x + threadIdx.x;
    long tot = (long)numK * CinPad * Cout;
    if (i >= tot) return;
    int co  = (int)(i % Cout);
    long t1 = i / Cout;
    int cip = (int)(t1 % CinPad);
    int k   = (int)(t1 / CinPad);
    float v = (cip < Cin) ? W[((long)k * Cin + cip) * Cout + co] : 0.0f;
    Wb[i] = (bf16)v;
}

// f32 activations [N][C] -> bf16 [N][Cpad], zero channel pad
__global__ void act_cvt(const float* __restrict__ X, bf16* __restrict__ Xb,
                        int N, int C, int Cpad)
{
    long i = (long)blockIdx.x * blockDim.x + threadIdx.x;
    long tot = (long)N * Cpad;
    if (i >= tot) return;
    int c = (int)(i % Cpad);
    long r = i / Cpad;
    float v = (c < C) ? X[r * C + c] : 0.0f;
    Xb[i] = (bf16)v;
}

// per-channel sum / sumsq (sums must be pre-zeroed); C divides grid*block
__global__ void bn_stats(const float* __restrict__ Y, int N, int C,
                         float* __restrict__ sums, float* __restrict__ sumsq)
{
    int tid = blockIdx.x * blockDim.x + threadIdx.x;
    int total = gridDim.x * blockDim.x;
    int c = tid % C;
    int step = total / C;
    float s = 0.0f, s2 = 0.0f;
    for (int r = tid / C; r < N; r += step) {
        float v = Y[(long)r * C + c];
        s += v; s2 += v * v;
    }
    atomicAdd(&sums[c], s);
    atomicAdd(&sumsq[c], s2);
}

// BN + ReLU, emit bf16 activations with channel padding
__global__ void bn_apply(const float* __restrict__ Y,
                         const float* __restrict__ sums, const float* __restrict__ sumsq,
                         const float* __restrict__ g, const float* __restrict__ b,
                         int N, int C, int Cpad, bf16* __restrict__ Xout, float eps)
{
    long i = (long)blockIdx.x * blockDim.x + threadIdx.x;
    long tot = (long)N * Cpad;
    if (i >= tot) return;
    int c = (int)(i % Cpad);
    long r = i / Cpad;
    float o = 0.0f;
    if (c < C) {
        float inv_n = 1.0f / (float)N;
        float mean = sums[c] * inv_n;
        float var  = sumsq[c] * inv_n - mean * mean;
        float v = (Y[r * C + c] - mean) * rsqrtf(var + eps) * g[c] + b[c];
        o = v > 0.0f ? v : 0.0f;
    }
    Xout[i] = (bf16)o;
}

// per-point BEV scatter: dense[bev_idx[n], :] += x[n, :128] @ K[bev_z[n]]
__global__ void bev_scatter(const bf16* __restrict__ X, const bf16* __restrict__ K,
                            const int* __restrict__ bidx, const int* __restrict__ bz,
                            float* __restrict__ dense, int N3)
{
    int n = blockIdx.x;
    if (n >= N3) return;
    int co = threadIdx.x;
    int z = bz[n];
    const bf16* x = X + (long)n * 128;
    const bf16* k = K + (long)z * 128 * CBEV + co;
    float s = 0.0f;
#pragma unroll 4
    for (int ci = 0; ci < 128; ++ci)
        s += (float)x[ci] * (float)k[(long)ci * CBEV];
    atomicAdd(&dense[(long)bidx[n] * CBEV + co], s);
}

__global__ void add_bias(float* __restrict__ dense, const float* __restrict__ bias, long tot)
{
    long i = (long)blockIdx.x * blockDim.x + threadIdx.x;
    if (i >= tot) return;
    dense[i] += bias[i % CBEV];
}

// BN + ReLU + NHWC->NCHW transpose into d_out
__global__ void bev_out(const float* __restrict__ dense,
                        const float* __restrict__ sums, const float* __restrict__ sumsq,
                        const float* __restrict__ g, const float* __restrict__ b,
                        float* __restrict__ out, int Nr)
{
    long i = (long)blockIdx.x * blockDim.x + threadIdx.x;
    long tot = (long)BATCH * CBEV * U3D * V3D;
    if (i >= tot) return;
    int v  = (int)(i % V3D);
    long t1 = i / V3D;
    int u  = (int)(t1 % U3D);
    long t2 = t1 / U3D;
    int c  = (int)(t2 % CBEV);
    int bb = (int)(t2 / CBEV);
    long r = ((long)bb * U3D + u) * V3D + v;
    float inv_n = 1.0f / (float)Nr;
    float mean = sums[c] * inv_n;
    float var  = sumsq[c] * inv_n - mean * mean;
    float x = (dense[r * CBEV + c] - mean) * rsqrtf(var + 1e-5f) * g[c] + b[c];
    out[i] = x > 0.0f ? x : 0.0f;
}

// ---------------------------------------------------------------------------
extern "C" void kernel_launch(void* const* d_in, const int* in_sizes, int n_in,
                              void* d_out, int out_size, void* d_ws, size_t ws_size,
                              hipStream_t stream)
{
    // Flattened input order (recursive dict-insertion order of setup_inputs):
    // 0 voxel_features; conv params as (W,g,b):
    //  stem: 1-3, 4-6 | stage1: 7-9,10-12,13-15 | stage2: 16-18,19-21,22-24,25-27
    //  stage3: 28-30,31-33,34-36,37-39 | stage4: 40-42,43-45,46-48 | out: 49-51
    //  bev: 52 kernel, 53 bias, 54 g, 55 b
    // 56 rb0, 57 rbs1, 58 rb1, 59 rbs2, 60 rb2, 61 rbs3, 62 rb3,
    // 63 bev_idx, 64 bev_z, 65 batch_size
    const float* vf = (const float*)d_in[0];
    const int N0 = in_sizes[0] / 4;
    const int N1 = in_sizes[57] / 27;
    const int N2 = in_sizes[59] / 27;
    const int N3 = in_sizes[63];

    const int* rb0  = (const int*)d_in[56];
    const int* rbs1 = (const int*)d_in[57];
    const int* rb1  = (const int*)d_in[58];
    const int* rbs2 = (const int*)d_in[59];
    const int* rb2  = (const int*)d_in[60];
    const int* rbs3 = (const int*)d_in[61];
    const int* rb3  = (const int*)d_in[62];
    const int* bev_idx = (const int*)d_in[63];
    const int* bev_z   = (const int*)d_in[64];

    size_t off = 0;
    auto carve = [&](size_t bytes) -> void* {
        off = (off + 255) & ~(size_t)255;
        void* p = (char*)d_ws + off;
        off += bytes;
        return p;
    };
    bf16*  actA  = (bf16*)carve((size_t)N0 * 128 * 2);
    bf16*  actB  = (bf16*)carve((size_t)N0 * 128 * 2);
    float* Y     = (float*)carve((size_t)N0 * 128 * 4);
    bf16*  Wb    = (bf16*)carve((size_t)27 * 128 * 128 * 2);
    float* sums  = (float*)carve(512 * 4);
    float* sumsq = sums + 256;
    float* dense = (float*)carve((size_t)BATCH * U3D * V3D * CBEV * 4);

    bf16* cur = actA;
    bf16* nxt = actB;

    auto wcvt = [&](int widx, int numK, int Cin, int CinPad, int Cout) {
        long tot = (long)numK * CinPad * Cout;
        conv_w_cvt<<<dim3((unsigned)((tot + 255) / 256)), 256, 0, stream>>>(
            (const float*)d_in[widx], Wb, numK, Cin, CinPad, Cout);
    };
    auto bnr = [&](int gidx, int bidx_, int N, int C, int CpadOut) {
        (void)hipMemsetAsync(sums, 0, 512 * 4, stream);
        bn_stats<<<512, 256, 0, stream>>>(Y, N, C, sums, sumsq);
        long tot = (long)N * CpadOut;
        bn_apply<<<dim3((unsigned)((tot + 255) / 256)), 256, 0, stream>>>(
            Y, sums, sumsq, (const float*)d_in[gidx], (const float*)d_in[bidx_],
            N, C, CpadOut, nxt, 1e-3f);
        bf16* t = cur; cur = nxt; nxt = t;
    };
    auto cdiv128 = [](int n) { return (unsigned)((n + 127) / 128); };

    // voxel features -> bf16 (N0 x 32, C=4)
    {
        long tot = (long)N0 * 32;
        act_cvt<<<dim3((unsigned)((tot + 255) / 256)), 256, 0, stream>>>(vf, actA, N0, 4, 32);
    }

    // ---- stem (4->16, 16->16 on rb0) ----
    wcvt(1, 27, 4, 32, 16);
    spconv_wmma<32, 16><<<dim3(cdiv128(N0)), 256, 0, stream>>>(cur, rb0, 27, Wb, Y, N0, N0);
    bnr(2, 3, N0, 16, 32);
    wcvt(4, 27, 16, 32, 16);
    spconv_wmma<32, 16><<<dim3(cdiv128(N0)), 256, 0, stream>>>(cur, rb0, 27, Wb, Y, N0, N0);
    bnr(5, 6, N0, 16, 32);

    // ---- stage1: 16->32 (rbs1, N0->N1), 2x 32->32 (rb1) ----
    wcvt(7, 27, 16, 32, 32);
    spconv_wmma<32, 32><<<dim3(cdiv128(N1)), 256, 0, stream>>>(cur, rbs1, 27, Wb, Y, N0, N1);
    bnr(8, 9, N1, 32, 32);
    {
        const int w[2] = {10, 13};
        for (int i = 0; i < 2; ++i) {
            wcvt(w[i], 27, 32, 32, 32);
            spconv_wmma<32, 32><<<dim3(cdiv128(N1)), 256, 0, stream>>>(cur, rb1, 27, Wb, Y, N1, N1);
            bnr(w[i] + 1, w[i] + 2, N1, 32, 32);
        }
    }

    // ---- stage2: 32->64 (rbs2, N1->N2), 3x 64->64 (rb2) ----
    wcvt(16, 27, 32, 32, 64);
    spconv_wmma<32, 64><<<dim3(cdiv128(N2)), 256, 0, stream>>>(cur, rbs2, 27, Wb, Y, N1, N2);
    bnr(17, 18, N2, 64, 64);
    {
        const int w[3] = {19, 22, 25};
        for (int i = 0; i < 3; ++i) {
            wcvt(w[i], 27, 64, 64, 64);
            spconv_wmma<64, 64><<<dim3(cdiv128(N2)), 256, 0, stream>>>(cur, rb2, 27, Wb, Y, N2, N2);
            bnr(w[i] + 1, w[i] + 2, N2, 64, 64);
        }
    }

    // ---- stage3: 64->128 (rbs3, N2->N3), 3x 128->128 (rb3) ----
    wcvt(28, 27, 64, 64, 128);
    spconv_wmma<64, 128><<<dim3(cdiv128(N3)), 256, 0, stream>>>(cur, rbs3, 27, Wb, Y, N2, N3);
    bnr(29, 30, N3, 128, 128);
    {
        const int w[6] = {31, 34, 37, 40, 43, 46};  // stage3[1..3] + stage4[0..2]
        for (int i = 0; i < 6; ++i) {
            wcvt(w[i], 27, 128, 128, 128);
            spconv_wmma<128, 128><<<dim3(cdiv128(N3)), 256, 0, stream>>>(cur, rb3, 27, Wb, Y, N3, N3);
            bnr(w[i] + 1, w[i] + 2, N3, 128, 128);
        }
    }

    // ---- out: 1x1 conv 128->128 (identity gather) ----
    wcvt(49, 1, 128, 128, 128);
    spconv_wmma<128, 128><<<dim3(cdiv128(N3)), 256, 0, stream>>>(cur, (const int*)nullptr, 1, Wb, Y, N3, N3);
    bnr(50, 51, N3, 128, 128);

    // ---- BEV: per-z linear 128->256, scatter-add, bias, BN, ReLU, NCHW ----
    wcvt(52, Z3D, 128, 128, CBEV);  // bev kernel -> bf16
    const long denseN = (long)BATCH * U3D * V3D;
    (void)hipMemsetAsync(dense, 0, (size_t)denseN * CBEV * 4, stream);
    bev_scatter<<<dim3((unsigned)N3), 256, 0, stream>>>(cur, Wb, bev_idx, bev_z, dense, N3);
    {
        long tot = denseN * CBEV;
        add_bias<<<dim3((unsigned)((tot + 255) / 256)), 256, 0, stream>>>(dense, (const float*)d_in[53], tot);
        (void)hipMemsetAsync(sums, 0, 512 * 4, stream);
        bn_stats<<<512, 256, 0, stream>>>(dense, (int)denseN, CBEV, sums, sumsq);
        bev_out<<<dim3((unsigned)((tot + 255) / 256)), 256, 0, stream>>>(
            dense, sums, sumsq, (const float*)d_in[54], (const float*)d_in[55],
            (float*)d_out, (int)denseN);
    }
}
